// BaudanauAttn_52570399703416
// MI455X (gfx1250) — matrix operand order repaired and checked by reading, not compiled, not run on hardware
//
#include <hip/hip_runtime.h>
#include <math.h>

// Bahdanau attention for MI455X (gfx1250), wave32 + WMMA bf16 + async LDS copy.
// B=32, L=1024, D=E2=1024.

#define Bn 32
#define Ln 1024
#define Dn 1024
#define En 1024

typedef __attribute__((ext_vector_type(16))) __bf16 v16bf;
typedef __attribute__((ext_vector_type(8)))  __bf16 v8bf;
typedef __attribute__((ext_vector_type(8)))  float  v8f;
typedef __attribute__((ext_vector_type(4)))  int    i32x4;

#if __has_builtin(__builtin_amdgcn_global_load_async_to_lds_b128) && \
    __has_builtin(__builtin_amdgcn_s_wait_asynccnt)
#define USE_ASYNC_LDS 1
#else
#define USE_ASYNC_LDS 0
#endif

typedef __attribute__((address_space(1))) i32x4 g_i32x4;
typedef __attribute__((address_space(3))) i32x4 l_i32x4;

__device__ __forceinline__ unsigned short f32_to_bf16(float f) {
    unsigned int u = __float_as_uint(f);
    unsigned int r = u + 0x7FFFu + ((u >> 16) & 1u);   // round-to-nearest-even
    return (unsigned short)(r >> 16);
}

// 16-byte global -> LDS copy (async on CDNA5: tracked by ASYNCcnt, bypasses VGPRs)
__device__ __forceinline__ void copy16_g2l(const unsigned short* g, unsigned short* l) {
#if USE_ASYNC_LDS
    __builtin_amdgcn_global_load_async_to_lds_b128(
        (g_i32x4*)const_cast<unsigned short*>(g),
        (l_i32x4*)l,
        0, 0);
#else
    *(uint4*)l = *(const uint4*)g;
#endif
}

__device__ __forceinline__ void wait_copies_landed() {
#if USE_ASYNC_LDS
    __builtin_amdgcn_s_wait_asynccnt(0);
#endif
}

// ---------------------------------------------------------------- convert f32 -> bf16
__global__ __launch_bounds__(256)
void cvt_bf16_kernel(const float* __restrict__ in, unsigned short* __restrict__ out,
                     long long n) {
    long long i = (long long)blockIdx.x * blockDim.x + threadIdx.x;
    long long stride = (long long)gridDim.x * blockDim.x;
    for (; i < n; i += stride) out[i] = f32_to_bf16(in[i]);
}

// ---------------------------------------------------------------- w1 = enc_last @ W1^T + b1
// grid (4, 32), block 256.  b = blockIdx.y, d = blockIdx.x*256 + tid.
// W1 (4MB) stays resident in the 192MB L2 across the 32 b-blocks.
__global__ __launch_bounds__(256)
void w1_kernel(const float* __restrict__ enc_out, const float* __restrict__ W1,
               const float* __restrict__ W1b, float* __restrict__ w1) {
    __shared__ float el[En];
    const int b = blockIdx.y;
    const float* encl = enc_out + ((size_t)b * Ln + (Ln - 1)) * En;
    for (int k = threadIdx.x; k < En; k += 256) el[k] = encl[k];
    __syncthreads();
    const int d = blockIdx.x * 256 + threadIdx.x;
    const float* wr = W1 + (size_t)d * En;
    float acc = W1b[d];
    for (int k = 0; k < En; ++k) acc = fmaf(el[k], wr[k], acc);
    w1[b * Dn + d] = acc;
}

// ---------------------------------------------------------------- WMMA GEMM
// MODE 0:  S[gm, d] = tanh(w1[b, d] + sum_k Hid[gm,k]*W2[d,k] + b2[d]) -> bf16
//          gm in [0, 32768), b = gm >> 10.  grid (8, 256, 1).
// MODE 1:  e[b, l, m] = sum_d Hid[b,l,d]*S[b,m,d] -> f32 into attn area.
//          grid (8, 8, 32), z = batch.
// Workgroup tile 128(M) x 128(N), K-step 32, 8 waves arranged 4(M) x 2(N),
// each wave: 32x64 = 2x4 wmma accumulators.  Double-buffered LDS with async
// global->LDS copies overlapping WMMA; one barrier per K-step.
template<int MODE>
__global__ __launch_bounds__(256)
void gemm_wmma_kernel(const unsigned short* __restrict__ Aall,  // bf16 rows x 1024
                      const unsigned short* __restrict__ Ball,  // bf16 rows x 1024
                      const float* __restrict__ w1,
                      const float* __restrict__ bias2,
                      unsigned short* __restrict__ Sout,
                      float* __restrict__ Eout) {
    __shared__ __align__(16) unsigned short As[2][128 * 32];
    __shared__ __align__(16) unsigned short Bs[2][128 * 32];

    const int K = 1024;
    const size_t batchOff = (MODE == 1) ? (size_t)blockIdx.z * (size_t)Ln * K : 0;
    const unsigned short* Ag = Aall + batchOff;
    const unsigned short* Bg = Ball + batchOff;

    const int mBase = blockIdx.y * 128;
    const int nBase = blockIdx.x * 128;

    const int lane = threadIdx.x & 31;
    const int wv   = threadIdx.x >> 5;      // 0..7
    const int mW   = (wv & 3) * 32;         // wave M offset inside tile
    const int nW   = (wv >> 2) * 64;        // wave N offset inside tile
    const int lrow = lane & 15;
    const int hi   = lane >> 4;             // lane group 0/1
    const int ka   = hi * 8;                // A-fragment K base (16-bit A layout)
    const int kb   = hi * 8;                // B-fragment K base (16-bit B layout)

    // Cooperative-copy slot assignment: 128 rows * 4 chunks(8 halfs) = 512 slots.
    const int s0 = threadIdx.x;
    const int s1 = threadIdx.x + 256;
    const int r0 = s0 >> 2, ko0 = (s0 & 3) * 8;
    const int r1 = s1 >> 2, ko1 = (s1 & 3) * 8;

    v8f acc[2][4];
#pragma unroll
    for (int mt = 0; mt < 2; ++mt)
#pragma unroll
        for (int nt = 0; nt < 4; ++nt)
            acc[mt][nt] = (v8f){0.f, 0.f, 0.f, 0.f, 0.f, 0.f, 0.f, 0.f};

    // Issue tile copy for k-chunk `kBase` into LDS buffer `buf`.
    auto issue_tile = [&](int buf, int kBase) {
        copy16_g2l(&Ag[(size_t)(mBase + r0) * K + kBase + ko0], &As[buf][r0 * 32 + ko0]);
        copy16_g2l(&Ag[(size_t)(mBase + r1) * K + kBase + ko1], &As[buf][r1 * 32 + ko1]);
        copy16_g2l(&Bg[(size_t)(nBase + r0) * K + kBase + ko0], &Bs[buf][r0 * 32 + ko0]);
        copy16_g2l(&Bg[(size_t)(nBase + r1) * K + kBase + ko1], &Bs[buf][r1 * 32 + ko1]);
    };

    issue_tile(0, 0);

    const int nSteps = K / 32;              // 32
    for (int i = 0; i < nSteps; ++i) {
        const int cur = i & 1;
        // This wave's outstanding copies (current buffer) have landed in LDS:
        wait_copies_landed();
        // Make them workgroup-visible; also guarantees everyone finished reading
        // the buffer we are about to overwrite below.
        __syncthreads();

        if (i + 1 < nSteps) issue_tile(cur ^ 1, (i + 1) * 32);

        // ---- compute on buffer `cur` ----
        v16bf af[2], bfv[4];
#pragma unroll
        for (int mt = 0; mt < 2; ++mt) {
            const int row = mW + mt * 16 + lrow;
            v8bf a0 = *(const v8bf*)&As[cur][row * 32 + ka];
            v8bf a1 = *(const v8bf*)&As[cur][row * 32 + ka + 16];
            af[mt] = __builtin_shufflevector(a0, a1, 0,1,2,3,4,5,6,7,8,9,10,11,12,13,14,15);
        }
#pragma unroll
        for (int nt = 0; nt < 4; ++nt) {
            const int row = nW + nt * 16 + lrow;
            v8bf b0 = *(const v8bf*)&Bs[cur][row * 32 + kb];
            v8bf b1 = *(const v8bf*)&Bs[cur][row * 32 + kb + 16];
            bfv[nt] = __builtin_shufflevector(b0, b1, 0,1,2,3,4,5,6,7,8,9,10,11,12,13,14,15);
        }
#pragma unroll
        for (int mt = 0; mt < 2; ++mt)
#pragma unroll
            for (int nt = 0; nt < 4; ++nt)
                acc[mt][nt] = __builtin_amdgcn_wmma_f32_16x16x32_bf16(
                    false, af[mt], false, bfv[nt], (short)0, acc[mt][nt], false, false);
    }

    // Epilogue.  C layout: VGPR r -> M = r + hi*8, N = lrow (16x16 f32 tile).
#pragma unroll
    for (int mt = 0; mt < 2; ++mt) {
#pragma unroll
        for (int nt = 0; nt < 4; ++nt) {
#pragma unroll
            for (int r = 0; r < 8; ++r) {
                const int ml = mW + mt * 16 + r + hi * 8;
                const int nl = nW + nt * 16 + lrow;
                const int gn = nBase + nl;
                if (MODE == 0) {
                    const size_t gm = (size_t)mBase + ml;       // 0..32767
                    const int b = (int)(gm >> 10);
                    float v = tanhf(acc[mt][nt][r] + w1[b * Dn + gn] + bias2[gn]);
                    Sout[gm * Dn + gn] = f32_to_bf16(v);
                } else {
                    const size_t l = (size_t)mBase + ml;        // row within batch
                    Eout[((size_t)blockIdx.z * Ln + l) * Ln + gn] = acc[mt][nt][r];
                }
            }
        }
    }
}

// ---------------------------------------------------------------- softmax over axis=1 (l)
// attn[b, :, m] normalized per column m.  grid (4, 32), block 256; thread = column m.
// Consecutive threads hit consecutive addresses -> coalesced strided sweeps.
__global__ __launch_bounds__(256)
void softmax_col_kernel(float* __restrict__ attn) {
    const int m = blockIdx.x * 256 + threadIdx.x;
    const int b = blockIdx.y;
    float* base = attn + (size_t)b * Ln * Ln + m;
    float mx = -INFINITY;
    for (int l = 0; l < Ln; ++l) mx = fmaxf(mx, base[(size_t)l * Ln]);
    float sum = 0.f;
    for (int l = 0; l < Ln; ++l) sum += expf(base[(size_t)l * Ln] - mx);
    const float inv = 1.0f / sum;
    for (int l = 0; l < Ln; ++l)
        base[(size_t)l * Ln] = expf(base[(size_t)l * Ln] - mx) * inv;
}

// ---------------------------------------------------------------- ct[b,l,e] = rowsum_m(attn[b,l,:]) * enc_last[b,e]
// grid (1024, 32), block 256.
__global__ __launch_bounds__(256)
void rowsum_ct_kernel(const float* __restrict__ attn, const float* __restrict__ enc_out,
                      float* __restrict__ ct) {
    const int l = blockIdx.x;
    const int b = blockIdx.y;
    const float* row = attn + ((size_t)b * Ln + l) * Ln;
    float p = 0.f;
    for (int m = threadIdx.x; m < Ln; m += 256) p += row[m];
    __shared__ float red[256];
    red[threadIdx.x] = p;
    __syncthreads();
    for (int s = 128; s > 0; s >>= 1) {
        if (threadIdx.x < s) red[threadIdx.x] += red[threadIdx.x + s];
        __syncthreads();
    }
    const float r = red[0];
    const float* el = enc_out + ((size_t)b * Ln + (Ln - 1)) * En;
    float* crow = ct + ((size_t)b * Ln + l) * En;
    for (int e = threadIdx.x; e < En; e += 256) crow[e] = r * el[e];
}

// ---------------------------------------------------------------- launch
extern "C" void kernel_launch(void* const* d_in, const int* in_sizes, int n_in,
                              void* d_out, int out_size, void* d_ws, size_t ws_size,
                              hipStream_t stream) {
    (void)in_sizes; (void)n_in; (void)out_size; (void)ws_size;
    const float* hid     = (const float*)d_in[0];  // (B,L,D)
    const float* enc_out = (const float*)d_in[1];  // (B,L,E2)
    /* d_in[2] = mask, unused by reference math */
    const float* W1w     = (const float*)d_in[3];  // (D,E2)
    const float* W1b     = (const float*)d_in[4];  // (D,)
    const float* W2w     = (const float*)d_in[5];  // (D,D)
    const float* W2b     = (const float*)d_in[6];  // (D,)

    float* ct   = (float*)d_out;                         // (B,L,E2)
    float* attn = ct + (size_t)Bn * Ln * En;             // (B,L,L)

    // Workspace layout (~130 MB)
    unsigned short* hid_bf = (unsigned short*)d_ws;                   // B*L*D bf16 = 64MB
    unsigned short* W2_bf  = hid_bf + (size_t)Bn * Ln * Dn;           // D*D bf16  = 2MB
    unsigned short* S_bf   = W2_bf + (size_t)Dn * Dn;                 // B*L*D bf16 = 64MB
    float*          w1     = (float*)(S_bf + (size_t)Bn * Ln * Dn);   // B*D f32 = 128KB

    cvt_bf16_kernel<<<2048, 256, 0, stream>>>(hid, hid_bf, (long long)Bn * Ln * Dn);
    cvt_bf16_kernel<<<256, 256, 0, stream>>>(W2w, W2_bf, (long long)Dn * Dn);

    w1_kernel<<<dim3(4, 32), 256, 0, stream>>>(enc_out, W1w, W1b, w1);

    // S = tanh(w1 + Hid @ W2^T + b2), bf16 out
    gemm_wmma_kernel<0><<<dim3(8, 256, 1), 256, 0, stream>>>(
        hid_bf, W2_bf, w1, W2b, S_bf, nullptr);

    // e = batched Hid @ S^T, f32 into attn region
    gemm_wmma_kernel<1><<<dim3(8, 8, 32), 256, 0, stream>>>(
        hid_bf, S_bf, nullptr, nullptr, nullptr, attn);

    softmax_col_kernel<<<dim3(4, 32), 256, 0, stream>>>(attn);

    rowsum_ct_kernel<<<dim3(1024, 32), 256, 0, stream>>>(attn, enc_out, ct);
}
